// RA_VQVAE_Encoder_2937757630620
// MI455X (gfx1250) — matrix operand order repaired and checked by reading, not compiled, not run on hardware
//
#include <hip/hip_runtime.h>
#include <hip/hip_bf16.h>
#include <math.h>

// ---------------------------------------------------------------------------
// RA-VQVAE encoder step for MI455X (gfx1250, wave32, WMMA bf16)
// ---------------------------------------------------------------------------

typedef __attribute__((ext_vector_type(16))) __bf16 v16bf;
typedef __attribute__((ext_vector_type(8)))  __bf16 v8bf;
typedef __attribute__((ext_vector_type(8)))  float  v8f;

#define N_ROWS   32768
#define DIM      256
#define KCB      2048
#define DECAYC   0.99f
#define EPSC     1e-5f
#define INV_TEMP 2.0f      // 1 / max(TEMP=0.5, 1e-3)
#define CCC      0.25f

// ---- helpers --------------------------------------------------------------

static __device__ inline void lds_fence() {
  // wave-private LDS producer/consumer sync (lockstep wave32, in-order DS)
  asm volatile("s_wait_dscnt 0" ::: "memory");
}

static __device__ inline unsigned short f2bf_bits(float f) {
  unsigned int u = __float_as_uint(f);
  unsigned int r = u + 0x7fffu + ((u >> 16) & 1u);   // round-to-nearest-even
  return (unsigned short)(r >> 16);
}

static __device__ inline __bf16 bits_to_bf(unsigned short u) {
  union { unsigned short u; __bf16 b; } x; x.u = u; return x.b;
}

static __device__ inline v16bf cat8(v8bf lo, v8bf hb) {
  v16bf r;
#pragma unroll
  for (int t = 0; t < 8; ++t) { r[t] = lo[t]; r[8 + t] = hb[t]; }
  return r;
}

static __device__ inline v8f zero8() {
  v8f z = {0.f, 0.f, 0.f, 0.f, 0.f, 0.f, 0.f, 0.f};
  return z;
}

static __device__ inline v8f wmma_bf16(v16bf a, v16bf b, v8f c) {
  return __builtin_amdgcn_wmma_f32_16x16x32_bf16(
      /*neg_a=*/false, a, /*neg_b=*/false, b,
      /*c_mod=*/(short)0, c, /*reuse_a=*/false, /*reuse_b=*/false);
}

// ---------------------------------------------------------------------------
// Kernel 1: codebook = emb @ proj_w^T + proj_b (exact f32), bf16 copies, |c|^2
// one block per codebook row k, 256 threads = D columns
// ---------------------------------------------------------------------------
__global__ __launch_bounds__(256)
void prep_kernel(const float* __restrict__ emb, const float* __restrict__ pw,
                 const float* __restrict__ pb,
                 float* __restrict__ cbF, float* __restrict__ c2,
                 __bf16* __restrict__ cbB, __bf16* __restrict__ cbT) {
  __shared__ float s_e[DIM];
  __shared__ float s_r[DIM];
  const int k = blockIdx.x;
  const int d = threadIdx.x;
  s_e[d] = emb[(size_t)k * DIM + d];
  __syncthreads();
  float acc = 0.f;
  const float* pwr = pw + (size_t)d * DIM;
  for (int j = 0; j < DIM; ++j) acc += s_e[j] * pwr[j];
  const float val = acc + pb[d];
  cbF[(size_t)k * DIM + d] = val;
  const __bf16 bv = bits_to_bf(f2bf_bits(val));
  cbB[(size_t)k * DIM + d] = bv;          // row-major  [K][D]  (GEMM1 B)
  cbT[(size_t)d * KCB + k] = bv;          // transposed [D][K]  (GEMM2 B)
  s_r[d] = val * val;
  __syncthreads();
  for (int off = 128; off > 0; off >>= 1) {
    if (d < off) s_r[d] += s_r[d + off];
    __syncthreads();
  }
  if (d == 0) c2[k] = s_r[0];
}

// ---------------------------------------------------------------------------
// Kernel 2: fused distances + argmin + softmax + soft quantize +
//           counts / embed_sum scatter + MSE partial sums.
// Block = 64 threads (2 waves). Wave 0 -> scRNA rows, wave 1 -> scATAC rows.
// Each block owns rows [blockIdx.x*16, +16).
// ---------------------------------------------------------------------------
__global__ __launch_bounds__(64)
void vq_main_kernel(const float* __restrict__ xRg, const float* __restrict__ xAg,
                    const float* __restrict__ cbF, const float* __restrict__ c2,
                    const __bf16* __restrict__ cbB, const __bf16* __restrict__ cbT,
                    float* __restrict__ countsR, float* __restrict__ countsA,
                    float* __restrict__ loss_acc, float* __restrict__ embed_sum,
                    float* __restrict__ outQ) {
  __shared__ __align__(16) unsigned short s_xbf[2][16][264]; // x tile, bf16
  __shared__ __align__(16) unsigned short s_pb[2][16][40];   // prob transpose buf
  __shared__ float s_x2[2][16];
  __shared__ float s_dmin[2][16];
  __shared__ int   s_idx[2][16];

  const int w    = threadIdx.x >> 5;      // modality wave
  const int lane = threadIdx.x & 31;
  const int arow = lane & 15;
  const int hi   = lane >> 4;
  const int rowBase = blockIdx.x * 16;

  const float* xs = (w == 0) ? xRg : xAg;
  const float* xo = (w == 0) ? xAg : xRg;
  float* cnt = (w == 0) ? countsR : countsA;
  float* q = outQ + (size_t)w * (size_t)N_ROWS * DIM;

  // ---- stage x tile (f32 -> bf16 LDS) + row squared norms -----------------
  for (int r = 0; r < 16; ++r) {
    const float* p = xs + (size_t)(rowBase + r) * DIM + lane * 8;
    float v[8]; float sq = 0.f;
#pragma unroll
    for (int t = 0; t < 8; ++t) { v[t] = p[t]; sq += v[t] * v[t]; }
#pragma unroll
    for (int t = 0; t < 8; ++t) s_xbf[w][r][lane * 8 + t] = f2bf_bits(v[t]);
#pragma unroll
    for (int off = 16; off > 0; off >>= 1) sq += __shfl_xor(sq, off);
    if (lane == 0) s_x2[w][r] = sq;
  }
  lds_fence();

  // ---- build A fragments (16x32 bf16 per 32-wide D chunk), kept resident --
  v16bf af[8];
#pragma unroll
  for (int j = 0; j < 8; ++j) {
    v8bf lo = *reinterpret_cast<const v8bf*>(&s_xbf[w][arow][32 * j + 16 * hi]);
    v8bf hb = *reinterpret_cast<const v8bf*>(&s_xbf[w][arow][32 * j + 16 * hi + 8]);
    af[j] = cat8(lo, hb);
  }
  float x2v[8];
#pragma unroll
  for (int i = 0; i < 8; ++i) x2v[i] = s_x2[w][i + 8 * hi];

  // ---- pass 1: distances, running min + argmin (branchless), double-buffered
  float m8[8]; int i8[8];
#pragma unroll
  for (int i = 0; i < 8; ++i) { m8[i] = 3.0e38f; i8[i] = 0; }

  const __bf16* bbase = cbB + (size_t)arow * DIM + 16 * hi;

  v16bf b0[8], b1[8];
#pragma unroll
  for (int j = 0; j < 8; ++j)
    b0[j] = *reinterpret_cast<const v16bf*>(bbase + 32 * j);

#pragma unroll 1
  for (int t = 0; t < 128; t += 2) {
    // prefetch tile t+1
    {
      const __bf16* bp = bbase + (size_t)(t + 1) * 16 * DIM;
#pragma unroll
      for (int j = 0; j < 8; ++j)
        b1[j] = *reinterpret_cast<const v16bf*>(bp + 32 * j);
    }
    // process tile t
    {
      v8f acc = zero8();
#pragma unroll
      for (int j = 0; j < 8; ++j) acc = wmma_bf16(af[j], b0[j], acc);
      const int col = t * 16 + arow;
      const float c2c = c2[col];
#pragma unroll
      for (int i = 0; i < 8; ++i) {
        const float dv = x2v[i] + c2c - 2.0f * acc[i];
        i8[i] = (dv < m8[i]) ? col : i8[i];
        m8[i] = fminf(m8[i], dv);
      }
    }
    // prefetch tile t+2
    if (t + 2 < 128) {
      const __bf16* bp = bbase + (size_t)(t + 2) * 16 * DIM;
#pragma unroll
      for (int j = 0; j < 8; ++j)
        b0[j] = *reinterpret_cast<const v16bf*>(bp + 32 * j);
    }
    // process tile t+1
    {
      v8f acc = zero8();
#pragma unroll
      for (int j = 0; j < 8; ++j) acc = wmma_bf16(af[j], b1[j], acc);
      const int col = (t + 1) * 16 + arow;
      const float c2c = c2[col];
#pragma unroll
      for (int i = 0; i < 8; ++i) {
        const float dv = x2v[i] + c2c - 2.0f * acc[i];
        i8[i] = (dv < m8[i]) ? col : i8[i];
        m8[i] = fminf(m8[i], dv);
      }
    }
  }

  // merge min/argmin across the 16 lanes of each half-wave
#pragma unroll
  for (int i = 0; i < 8; ++i) {
    float m = m8[i]; int id = i8[i];
#pragma unroll
    for (int off = 1; off < 16; off <<= 1) {
      const float m2 = __shfl_xor(m, off);
      const int  id2 = __shfl_xor(id, off);
      id = (m2 < m) ? id2 : id;
      m = fminf(m, m2);
    }
    m8[i] = m; i8[i] = id;
  }
  if (arow == 0) {
#pragma unroll
    for (int i = 0; i < 8; ++i) {
      s_dmin[w][i + 8 * hi] = m8[i];
      s_idx [w][i + 8 * hi] = i8[i];
    }
  }
  lds_fence();
  float dmin8[8];
#pragma unroll
  for (int i = 0; i < 8; ++i) dmin8[i] = s_dmin[w][i + 8 * hi];

  // ---- pass 2: recompute dists -> unnormalized probs -> softq & row sums --
  v8f sq16[16];
#pragma unroll
  for (int dt = 0; dt < 16; ++dt) sq16[dt] = zero8();
  float ssum8[8];
#pragma unroll
  for (int i = 0; i < 8; ++i) ssum8[i] = 0.f;

#pragma unroll 1
  for (int c = 0; c < 64; ++c) {
    const int colBase = c * 32;
#pragma unroll
    for (int sub = 0; sub < 2; ++sub) {
      const int col = colBase + sub * 16 + arow;
      const __bf16* bp = cbB + (size_t)col * DIM + 16 * hi;
      v16bf bt[8];
#pragma unroll
      for (int j = 0; j < 8; ++j)
        bt[j] = *reinterpret_cast<const v16bf*>(bp + 32 * j);
      v8f acc = zero8();
#pragma unroll
      for (int j = 0; j < 8; ++j) acc = wmma_bf16(af[j], bt[j], acc);
      const float c2c = c2[col];
#pragma unroll
      for (int i = 0; i < 8; ++i) {
        const float dv = x2v[i] + c2c - 2.0f * acc[i];
        const float p  = __expf((dmin8[i] - dv) * INV_TEMP);  // <= 1
        ssum8[i] += p;
        s_pb[w][i + 8 * hi][sub * 16 + arow] = f2bf_bits(p);
      }
    }
    lds_fence();
    v8bf lo = *reinterpret_cast<const v8bf*>(&s_pb[w][arow][16 * hi]);
    v8bf hb = *reinterpret_cast<const v8bf*>(&s_pb[w][arow][16 * hi + 8]);
    v16bf pa = cat8(lo, hb);
#pragma unroll
    for (int dt = 0; dt < 16; ++dt) {
      const int n = dt * 16 + arow;
      v16bf b = *reinterpret_cast<const v16bf*>(cbT + (size_t)n * KCB + colBase + 16 * hi);
      sq16[dt] = wmma_bf16(pa, b, sq16[dt]);
    }
  }

  // row softmax sums -> reciprocal (normalization folded into output scaling)
  float inv8[8];
#pragma unroll
  for (int i = 0; i < 8; ++i) {
    float s = ssum8[i];
#pragma unroll
    for (int off = 1; off < 16; off <<= 1) s += __shfl_xor(s, off);
    inv8[i] = 1.0f / s;
  }

  // ---- write quantized output: x + (softq - x) ----------------------------
#pragma unroll
  for (int dt = 0; dt < 16; ++dt) {
#pragma unroll
    for (int i = 0; i < 8; ++i) {
      const int row  = rowBase + i + 8 * hi;
      const int colD = dt * 16 + arow;
      const size_t gi = (size_t)row * DIM + colD;
      const float xv = xs[gi];
      q[gi] = xv + (sq16[dt][i] * inv8[i] - xv);
    }
  }

  // ---- counts, embed_sum scatter, loss partial sums -----------------------
  if (lane < 16) atomicAdd(&cnt[s_idx[w][lane]], 1.0f);

  float lself = 0.f, lcross = 0.f;
  for (int r = 0; r < 16; ++r) {
    const int k = s_idx[w][r];
    const size_t xro = (size_t)(rowBase + r) * DIM;
    const size_t cbo = (size_t)k * DIM;
    for (int col = lane; col < DIM; col += 32) {
      const float cv = cbF[cbo + col];
      const float a  = xs[xro + col];
      const float b  = xo[xro + col];
      atomicAdd(&embed_sum[cbo + col], a);
      const float d1 = cv - a; lself  += d1 * d1;   // hard_self vs x_self
      const float d2 = cv - b; lcross += d2 * d2;   // hard_self vs x_other
    }
  }
#pragma unroll
  for (int off = 16; off > 0; off >>= 1) {
    lself  += __shfl_xor(lself,  off);
    lcross += __shfl_xor(lcross, off);
  }
  if (lane == 0) {
    atomicAdd(&loss_acc[w], lself);        // 0: ssRR, 1: ssAA
    atomicAdd(&loss_acc[2 + w], lcross);   // 2: ssRA (hardR vs xA), 3: ssAR
  }
}

// ---------------------------------------------------------------------------
// Kernel 3: EMA updates -> new_ema_w, new_cluster_size, new_embedding
// ---------------------------------------------------------------------------
__global__ __launch_bounds__(256)
void fin1_kernel(const float* __restrict__ emaw, const float* __restrict__ emacs,
                 const float* __restrict__ countsR, const float* __restrict__ countsA,
                 const float* __restrict__ embed_sum,
                 float* __restrict__ outEmb, float* __restrict__ outCS,
                 float* __restrict__ outEmaW) {
  const size_t i = (size_t)blockIdx.x * 256 + threadIdx.x;
  const int k = (int)(i >> 8);
  const int d = (int)(i & 255);
  const float nw  = emaw[i] * DECAYC + (1.0f - DECAYC) * embed_sum[i];
  const float ncs = emacs[k] * DECAYC + (1.0f - DECAYC) * (countsR[k] + countsA[k]);
  outEmaW[i] = nw;
  outEmb[i]  = nw / fmaxf(ncs, EPSC);
  if (d == 0) outCS[k] = ncs;
}

// ---------------------------------------------------------------------------
// Kernel 4: perplexities + scalar losses
// ---------------------------------------------------------------------------
__global__ __launch_bounds__(256)
void fin2_kernel(const float* __restrict__ countsR, const float* __restrict__ countsA,
                 const float* __restrict__ loss_acc, float* __restrict__ outScal) {
  __shared__ float sR[256], sA[256];
  float aR = 0.f, aA = 0.f;
  for (int k = threadIdx.x; k < KCB; k += 256) {
    const float r = countsR[k] * (1.0f / (float)N_ROWS);
    const float a = countsA[k] * (1.0f / (float)N_ROWS);
    aR += r * __logf(r + 1e-10f);
    aA += a * __logf(a + 1e-10f);
  }
  sR[threadIdx.x] = aR; sA[threadIdx.x] = aA;
  __syncthreads();
  for (int off = 128; off > 0; off >>= 1) {
    if (threadIdx.x < off) {
      sR[threadIdx.x] += sR[threadIdx.x + off];
      sA[threadIdx.x] += sA[threadIdx.x + off];
    }
    __syncthreads();
  }
  if (threadIdx.x == 0) {
    const float invND = 1.0f / ((float)N_ROWS * (float)DIM);
    const float mRR = loss_acc[0] * invND;   // mse(hardR, xR)
    const float mAA = loss_acc[1] * invND;   // mse(hardA, xA)
    const float mRA = loss_acc[2] * invND;   // mse(hardR, xA)
    const float mAR = loss_acc[3] * invND;   // mse(hardA, xR)
    outScal[0] = 2.0f * CCC * mRR;                               // scRNA_loss
    const float fwd = mAA + mRR + 0.5f * mAR + 0.5f * mRA;
    outScal[1] = 2.0f * CCC * mAA + CCC * fwd;                   // scATAC_loss
    outScal[2] = __expf(-sR[0]);                                 // perpR
    outScal[3] = __expf(-sA[0]);                                 // perpA
  }
}

// ---------------------------------------------------------------------------
extern "C" void kernel_launch(void* const* d_in, const int* in_sizes, int n_in,
                              void* d_out, int out_size, void* d_ws, size_t ws_size,
                              hipStream_t stream) {
  const float* xR    = (const float*)d_in[0];
  const float* xA    = (const float*)d_in[1];
  const float* emb   = (const float*)d_in[2];
  const float* pw    = (const float*)d_in[3];
  const float* pb    = (const float*)d_in[4];
  const float* emacs = (const float*)d_in[5];
  const float* emaw  = (const float*)d_in[6];

  const size_t ND = (size_t)N_ROWS * DIM;   // 8388608
  const size_t KD = (size_t)KCB * DIM;      // 524288

  float* out      = (float*)d_out;
  float* outScal  = out + 2 * ND;           // lossR, lossA, perpR, perpA
  float* outEmb   = out + 2 * ND + 4;
  float* outCS    = outEmb + KD;
  float* outEmaW  = outCS + KCB;

  char* w = (char*)d_ws;
  float*  cbF = (float*)w;   w += KD * sizeof(float);
  float*  c2  = (float*)w;   w += KCB * sizeof(float);
  __bf16* cbB = (__bf16*)w;  w += KD * sizeof(__bf16);
  __bf16* cbT = (__bf16*)w;  w += KD * sizeof(__bf16);
  float* countsR   = (float*)w;
  float* countsA   = countsR + KCB;
  float* loss_acc  = countsA + KCB;
  float* embed_sum = loss_acc + 8;

  // zero accumulators (counts + losses + embed_sum are contiguous)
  hipMemsetAsync(countsR, 0, (2 * KCB + 8 + KD) * sizeof(float), stream);

  prep_kernel<<<KCB, 256, 0, stream>>>(emb, pw, pb, cbF, c2, cbB, cbT);
  vq_main_kernel<<<N_ROWS / 16, 64, 0, stream>>>(xR, xA, cbF, c2, cbB, cbT,
                                                 countsR, countsA, loss_acc,
                                                 embed_sum, out);
  fin1_kernel<<<(unsigned)(KD / 256), 256, 0, stream>>>(emaw, emacs, countsR,
                                                        countsA, embed_sum,
                                                        outEmb, outCS, outEmaW);
  fin2_kernel<<<1, 256, 0, stream>>>(countsR, countsA, loss_acc, outScal);
}
